// ModelClassifier_69844758167924
// MI455X (gfx1250) — matrix operand order repaired and checked by reading, not compiled, not run on hardware
//
#include <hip/hip_runtime.h>
#include <cstdint>
#include <cstddef>

// ---------------------------------------------------------------------------
// ResNet-50 forward + per-make expert head for MI455X (gfx1250, wave32, WMMA)
// All 1x1 / 3x3 convs run as implicit GEMM on v_wmma_f32_16x16x32_f16.
// Block tile: 64 OC x 32 px, 4 waves, K chunked by 64 -> 4 WMMA per chunk.
// ---------------------------------------------------------------------------

typedef _Float16 half_t;
typedef __attribute__((ext_vector_type(16))) _Float16 v16h;
typedef __attribute__((ext_vector_type(8)))  float    v8f;

union FragA { v16h v; unsigned int u[8]; };
union FragC { v8f  v; float        f[8]; };

// --- fp32 OIHW weights -> f16 [OC][IC*KH*KW] with BN gamma folded in --------
__global__ void pack_weights_kernel(const float* __restrict__ w,
                                    const float* __restrict__ g,
                                    half_t* __restrict__ out,
                                    int total, int ick) {
  int i = blockIdx.x * 256 + threadIdx.x;
  if (i < total) {
    int oc = i / ick;
    out[i] = (half_t)(w[i] * g[oc]);
  }
}

// --- conv1: 7x7 s2 p3, 3->64 ch, fused BN + ReLU, fp32 in -> f16 out --------
__global__ void conv7x7_bn_relu_kernel(const float* __restrict__ x,
                                       const float* __restrict__ w,
                                       const float* __restrict__ gamma,
                                       const float* __restrict__ beta,
                                       half_t* __restrict__ out) {
  int t = blockIdx.x * 256 + threadIdx.x;           // exactly 32*64*112*112
  int ox = t % 112;
  int oy = (t / 112) % 112;
  int oc = (t / 12544) % 64;
  int n  = t / 802816;
  float acc = 0.f;
  for (int ic = 0; ic < 3; ++ic)
    for (int ky = 0; ky < 7; ++ky) {
      int iy = oy * 2 - 3 + ky;
      if ((unsigned)iy >= 224u) continue;
      for (int kx = 0; kx < 7; ++kx) {
        int ix = ox * 2 - 3 + kx;
        if ((unsigned)ix >= 224u) continue;
        acc += x[(((size_t)n * 3 + ic) * 224 + iy) * 224 + ix] *
               w[((oc * 3 + ic) * 7 + ky) * 7 + kx];
      }
    }
  float v = acc * gamma[oc] + beta[oc];
  out[t] = (half_t)(v > 0.f ? v : 0.f);
}

// --- maxpool 3x3 s2 p1 : [32,64,112,112] -> [32,64,56,56] (f16) -------------
__global__ void maxpool_kernel(const half_t* __restrict__ in,
                               half_t* __restrict__ out) {
  int t = blockIdx.x * 256 + threadIdx.x;           // exactly 32*64*56*56
  int ox = t % 56;
  int oy = (t / 56) % 56;
  int c  = (t / 3136) % 64;
  int n  = t / 200704;
  float m = -3.4e38f;
  for (int ky = 0; ky < 3; ++ky) {
    int iy = oy * 2 - 1 + ky;
    if ((unsigned)iy >= 112u) continue;
    for (int kx = 0; kx < 3; ++kx) {
      int ix = ox * 2 - 1 + kx;
      if ((unsigned)ix >= 112u) continue;
      float v = (float)in[(((size_t)n * 64 + c) * 112 + iy) * 112 + ix];
      m = v > m ? v : m;
    }
  }
  out[t] = (half_t)m;
}

// --- generic implicit-GEMM conv on WMMA -------------------------------------
// Out[OC,P] = W[OC,K] x In[K,P], K = IC*KHW (mult of 64), P = N*Hout*Wout
// (mult of 32). 128 threads = 4 waves. Block tile = 64 OC x 32 px; each wave
// holds 2 f32 accumulators (16 OC x 2x16 px) and issues 4 WMMA per K-chunk
// of 64, reusing each A fragment across both pixel tiles.
// Epilogue: + bias, + optional residual (may alias `out`), optional ReLU.
template <int KHW>
__global__ __launch_bounds__(128)
void conv_wmma(const half_t* __restrict__ in,
               const half_t* __restrict__ wG,
               const float*  __restrict__ bias,
               const half_t* res,          // nullable; may alias out
               half_t*       out,          // NCHW f16
               int IC, int Hin, int Win,
               int OC, int Hout, int Wout,
               int stride, int relu) {
  const int K      = IC * KHW;
  const int ocBase = blockIdx.x * 64;
  const int pBase  = blockIdx.y * 32;

  const int lane = threadIdx.x & 31;
  const int wv   = threadIdx.x >> 5;   // wave 0..3 -> 16 OC rows each
  const int lm   = lane & 15;
  const int lh   = lane >> 4;

  __shared__ __align__(16) half_t sW[64][68];  // 64 oc x 64 k (+pad, 34 banks)
  __shared__ __align__(16) half_t sI[32][68];  // 32 px x 64 k (+pad)

  v8f acc0 = {0.f, 0.f, 0.f, 0.f, 0.f, 0.f, 0.f, 0.f};
  v8f acc1 = {0.f, 0.f, 0.f, 0.f, 0.f, 0.f, 0.f, 0.f};

  const int HWo = Hout * Wout;

  // ---- per-thread staging coordinates (hoisted out of the K loop) ---------
  // weight staging: thread -> (row, 32-elem half of the 64-wide chunk)
  const int wRow  = threadIdx.x >> 1;
  const int wHalf = (threadIdx.x & 1) * 32;
  // input staging: thread -> (pixel, 16-elem quarter of the 64-wide chunk)
  const int gP   = threadIdx.x >> 2;        // 0..31
  const int gK0  = (threadIdx.x & 3) * 16;  // 0,16,32,48
  const int gPg  = pBase + gP;
  const int gN   = gPg / HWo;
  const int gRem = gPg % HWo;
  const int gOy  = gRem / Wout, gOx = gRem % Wout;

  for (int kc = 0; kc < K; kc += 64) {
    // ---- stage weight tile: 64 rows x 64 k, 16B-vectorized ---------------
    {
      const uint4* src =
          (const uint4*)(wG + (size_t)(ocBase + wRow) * K + kc + wHalf);
      uint4 q0 = src[0], q1 = src[1], q2 = src[2], q3 = src[3];
      if (kc + 64 < K)
        __builtin_prefetch(wG + (size_t)(ocBase + wRow) * K + kc + 64, 0, 0);
      uint2* dst = (uint2*)&sW[wRow][wHalf];
      dst[0] = make_uint2(q0.x, q0.y); dst[1] = make_uint2(q0.z, q0.w);
      dst[2] = make_uint2(q1.x, q1.y); dst[3] = make_uint2(q1.z, q1.w);
      dst[4] = make_uint2(q2.x, q2.y); dst[5] = make_uint2(q2.z, q2.w);
      dst[6] = make_uint2(q3.x, q3.y); dst[7] = make_uint2(q3.z, q3.w);
    }
    // ---- gather input tile: 32 px x 64 k (implicit im2col) ---------------
    {
      for (int j = 0; j < 16; ++j) {
        int kk = gK0 + j;
        int k  = kc + kk;
        half_t val = (half_t)0.f;
        if (KHW == 1) {
          val = in[(((size_t)gN * IC + k) * Hin + gOy * stride) * Win +
                   gOx * stride];
        } else {  // 3x3, pad 1
          int ic = k / 9;
          int r  = k % 9;
          int ky = r / 3, kx = r % 3;
          int iy = gOy * stride - 1 + ky;
          int ix = gOx * stride - 1 + kx;
          if ((unsigned)iy < (unsigned)Hin && (unsigned)ix < (unsigned)Win)
            val = in[(((size_t)gN * IC + ic) * Hin + iy) * Win + ix];
        }
        sI[gP][kk] = val;
      }
    }
    __syncthreads();

    // ---- fragments per ISA 16-bit A(16x32)/B(32x16) layouts --------------
    FragA A0, A1, B00, B01, B10, B11;
    const unsigned int* pa  = (const unsigned int*)&sW[wv * 16 + lm][0];
    const unsigned int* pb0 = (const unsigned int*)&sI[lm][0];
    const unsigned int* pb1 = (const unsigned int*)&sI[16 + lm][0];
#pragma unroll
    for (int j = 0; j < 8; ++j) {
      int ai = ((j >> 2) << 3) + (lh << 2) + (j & 3);  // A k/2 index
      int bi = ((j >> 2) << 3) + (lh << 2) + (j & 3);  // B k/2 index
      A0.u[j]  = pa[ai];        A1.u[j]  = pa[16 + ai];
      B00.u[j] = pb0[bi];       B01.u[j] = pb0[16 + bi];
      B10.u[j] = pb1[bi];       B11.u[j] = pb1[16 + bi];
    }
    acc0 = __builtin_amdgcn_wmma_f32_16x16x32_f16(
        false, A0.v, false, B00.v, (short)0, acc0, false, false);
    acc1 = __builtin_amdgcn_wmma_f32_16x16x32_f16(
        false, A0.v, false, B10.v, (short)0, acc1, false, false);
    acc0 = __builtin_amdgcn_wmma_f32_16x16x32_f16(
        false, A1.v, false, B01.v, (short)0, acc0, false, false);
    acc1 = __builtin_amdgcn_wmma_f32_16x16x32_f16(
        false, A1.v, false, B11.v, (short)0, acc1, false, false);
    __syncthreads();
  }

  // ---- epilogue: bias (+residual) (+ReLU), f32 -> f16 store ---------------
#pragma unroll
  for (int t = 0; t < 2; ++t) {
    int pg  = pBase + t * 16 + lm;
    int n   = pg / HWo;
    int rem = pg % HWo;
    int oy  = rem / Wout, ox = rem % Wout;
    FragC C; C.v = t ? acc1 : acc0;
#pragma unroll
    for (int r = 0; r < 8; ++r) {
      int oc = ocBase + wv * 16 + lh * 8 + r;   // C: VGPR r -> M = r + 8*hi
      float  val = C.f[r] + bias[oc];
      size_t oi  = (((size_t)n * OC + oc) * Hout + oy) * Wout + ox;
      if (res) val += (float)res[oi];
      if (relu) val = val > 0.f ? val : 0.f;
      out[oi] = (half_t)val;
    }
  }
}

// --- global average pool: [32,2048,7,7] f16 -> [32,2048] f32 ----------------
__global__ void avgpool_kernel(const half_t* __restrict__ in,
                               float* __restrict__ feats) {
  int t = blockIdx.x * 256 + threadIdx.x;  // 65536
  const half_t* p = in + (size_t)t * 49;
  float s = 0.f;
  for (int i = 0; i < 49; ++i) s += (float)p[i];
  feats[t] = s * (1.0f / 49.0f);
}

// --- per-sample expert head: out[b,m] = [feats|emb] . W[e] + b[e] -----------
__global__ void head_kernel(const float* __restrict__ feats,
                            const float* __restrict__ emb,
                            const float* __restrict__ ew,
                            const float* __restrict__ eb,
                            const int*   __restrict__ make_ids,
                            float* __restrict__ out) {
  int b = blockIdx.x;
  int m = threadIdx.x & 31;
  int g = threadIdx.x >> 5;  // 8 partial groups
  int e = make_ids[b];
  const float* w = ew + (size_t)e * 2176 * 32;
  float acc = 0.f;
  for (int d = g; d < 2048; d += 8) acc += feats[b * 2048 + d] * w[d * 32 + m];
  for (int d = g; d < 128;  d += 8) acc += emb[e * 128 + d] * w[(2048 + d) * 32 + m];
  __shared__ float red[8][32];
  red[g][m] = acc;
  __syncthreads();
  if (g == 0) {
    float s = acc;
    for (int i = 1; i < 8; ++i) s += red[i][m];
    out[b * 32 + m] = s + eb[e * 32 + m];
  }
}

// ---------------------------------------------------------------------------
extern "C" void kernel_launch(void* const* d_in, const int* in_sizes, int n_in,
                              void* d_out, int out_size, void* d_ws,
                              size_t ws_size, hipStream_t stream) {
  (void)in_sizes; (void)n_in; (void)out_size; (void)ws_size;

  const float* x        = (const float*)d_in[0];
  const int*   make_ids = (const int*)d_in[1];
  const float* conv1w   = (const float*)d_in[2];
  const float* g1       = (const float*)d_in[3];
  const float* b1       = (const float*)d_in[4];

  struct Layer { const float *w, *g, *b; int ic, oc, khw; size_t off; };
  struct Blk   { int l1, l2, l3, ld; int cin, mid, cout, stride; };
  Layer L[53]; int nl = 0;
  Blk   blks[16]; int nb = 0;

  const int stage_mid[4] = {64, 128, 256, 512};
  const int stage_nb[4]  = {3, 4, 6, 3};

  int idx = 5;
  int cin = 64;
  size_t woff = 0;
  for (int s = 0; s < 4; ++s) {
    int mid = stage_mid[s], outc = mid * 4;
    for (int i = 0; i < stage_nb[s]; ++i) {
      Blk bk;
      bk.cin = cin; bk.mid = mid; bk.cout = outc;
      bk.stride = (i == 0 && s > 0) ? 2 : 1;
      L[nl] = {(const float*)d_in[idx + 0], (const float*)d_in[idx + 1],
               (const float*)d_in[idx + 2], cin, mid, 1, woff};
      woff += (size_t)mid * cin; bk.l1 = nl++;
      L[nl] = {(const float*)d_in[idx + 3], (const float*)d_in[idx + 4],
               (const float*)d_in[idx + 5], mid, mid, 9, woff};
      woff += (size_t)mid * mid * 9; bk.l2 = nl++;
      L[nl] = {(const float*)d_in[idx + 6], (const float*)d_in[idx + 7],
               (const float*)d_in[idx + 8], mid, outc, 1, woff};
      woff += (size_t)outc * mid; bk.l3 = nl++;
      idx += 9;
      if (i == 0) {  // downsample wd,gd,bd
        L[nl] = {(const float*)d_in[idx + 0], (const float*)d_in[idx + 1],
                 (const float*)d_in[idx + 2], cin, outc, 1, woff};
        woff += (size_t)outc * cin; bk.ld = nl++;
        idx += 3;
      } else {
        bk.ld = -1;
      }
      blks[nb++] = bk;
      cin = outc;
    }
  }
  const float* emb   = (const float*)d_in[idx++];
  const float* ew    = (const float*)d_in[idx++];
  const float* ebias = (const float*)d_in[idx++];

  // ---- workspace carve ----------------------------------------------------
  char* ws = (char*)d_ws;
  size_t o = 0;
  auto carve = [&](size_t bytes) -> void* {
    void* p = ws + o;
    o += (bytes + 255) & ~(size_t)255;
    return p;
  };
  half_t* wpack = (half_t*)carve(woff * sizeof(half_t));       // ~47 MB
  half_t* bufA  = (half_t*)carve((size_t)25690112 * 2);        // 51 MB
  half_t* bufB  = (half_t*)carve((size_t)25690112 * 2);        // 51 MB
  half_t* y1    = (half_t*)carve((size_t)12845056 * 2);        // 26 MB
  half_t* y2    = (half_t*)carve((size_t)12845056 * 2);        // 26 MB
  float*  feats = (float*)carve((size_t)65536 * 4);

  // ---- pack all conv weights (BN gamma folded) ----------------------------
  for (int l = 0; l < nl; ++l) {
    int total = L[l].oc * L[l].ic * L[l].khw;
    pack_weights_kernel<<<(total + 255) / 256, 256, 0, stream>>>(
        L[l].w, L[l].g, wpack + L[l].off, total, L[l].ic * L[l].khw);
  }

  // ---- stem ---------------------------------------------------------------
  conv7x7_bn_relu_kernel<<<100352, 256, 0, stream>>>(x, conv1w, g1, b1, bufA);
  maxpool_kernel<<<25088, 256, 0, stream>>>(bufA, bufB);

  // ---- bottleneck stages --------------------------------------------------
  half_t* h     = bufB;
  half_t* other = bufA;
  int H = 56, W = 56;
  for (int i = 0; i < nb; ++i) {
    const Blk& bk = blks[i];
    int Ho = H / bk.stride, Wo = W / bk.stride;
    const Layer &A1 = L[bk.l1], &A2 = L[bk.l2], &A3 = L[bk.l3];

    // y1 = relu(bn(conv1x1(h)))
    {
      int P = 32 * H * W;
      conv_wmma<1><<<dim3(bk.mid / 64, P / 32), 128, 0, stream>>>(
          h, wpack + A1.off, A1.b, nullptr, y1,
          bk.cin, H, W, bk.mid, H, W, 1, 1);
    }
    // y2 = relu(bn(conv3x3(y1, stride)))
    {
      int P = 32 * Ho * Wo;
      conv_wmma<9><<<dim3(bk.mid / 64, P / 32), 128, 0, stream>>>(
          y1, wpack + A2.off, A2.b, nullptr, y2,
          bk.mid, H, W, bk.mid, Ho, Wo, bk.stride, 1);
    }
    // residual: downsample into `other`, or identity `h`
    const half_t* res;
    if (bk.ld >= 0) {
      const Layer& AD = L[bk.ld];
      int P = 32 * Ho * Wo;
      conv_wmma<1><<<dim3(bk.cout / 64, P / 32), 128, 0, stream>>>(
          h, wpack + AD.off, AD.b, nullptr, other,
          bk.cin, H, W, bk.cout, Ho, Wo, bk.stride, 0);
      res = other;  // aliases output of the next conv (element-wise safe)
    } else {
      res = h;
    }
    // h' = relu(bn(conv1x1(y2)) + res)
    {
      int P = 32 * Ho * Wo;
      conv_wmma<1><<<dim3(bk.cout / 64, P / 32), 128, 0, stream>>>(
          y2, wpack + A3.off, A3.b, res, other,
          bk.mid, Ho, Wo, bk.cout, Ho, Wo, 1, 1);
    }
    half_t* t = h; h = other; other = t;
    H = Ho; W = Wo;
  }

  // ---- head ---------------------------------------------------------------
  avgpool_kernel<<<256, 256, 0, stream>>>(h, feats);
  head_kernel<<<32, 256, 0, stream>>>(feats, emb, ew, ebias, make_ids,
                                      (float*)d_out);
}
